// Matrix_65584150610521
// MI455X (gfx1250) — matrix operand order repaired and checked by reading, not compiled, not run on hardware
//
#include <hip/hip_runtime.h>
#include <hip/hip_bf16.h>
#include <math.h>

typedef __attribute__((ext_vector_type(16))) _Float16 v16h;
typedef __attribute__((ext_vector_type(8)))  _Float16 v8h;
typedef __attribute__((ext_vector_type(4)))  _Float16 v4h;
typedef __attribute__((ext_vector_type(8)))  float    v8f;
typedef int v4i_b __attribute__((vector_size(16)));   // matches builtin param type

#define SDIM 20
#define HDIM 256
#define HPAD 264   // padded LDS stride in halves (16B-aligned, bank-spread)

#if defined(__gfx1250__) && __has_builtin(__builtin_amdgcn_global_load_async_to_lds_b128) && \
    __has_builtin(__builtin_amdgcn_s_wait_asynccnt)
#define HAS_ASYNC_LDS 1
typedef __attribute__((address_space(1))) v4i_b* gas_v4i;
typedef __attribute__((address_space(3))) v4i_b* las_v4i;
#endif

// ---------------------------------------------------------------------------
// Prep 1: A[bs,p,h] = xf . W1[:258],  B[bs,p,h] = xf . W1[258:]   (fp32)
// xf[bs,p,c] = x[bs,c,p] for c<256 ; c=256 -> i/4 ; c=257 -> j/4 (p = i*4+j)
// ---------------------------------------------------------------------------
__global__ void prep_ab(const float* __restrict__ x, const float* __restrict__ W1,
                        float* __restrict__ A, float* __restrict__ B) {
    const int row = blockIdx.x;        // 0..639 : (b*S+s)*16 + p
    const int h   = threadIdx.x;       // 0..255
    const int p   = row & 15;
    const int bs  = row >> 4;
    const float ci = (float)(p >> 2) * 0.25f;
    const float cj = (float)(p & 3) * 0.25f;
    const float* xp = x + (size_t)bs * 256 * 16 + p;   // stride 16 over c
    float sa = 0.f, sb = 0.f;
    for (int c = 0; c < 256; ++c) {
        const float xv = xp[c * 16];
        sa += xv * W1[c * 256 + h];
        sb += xv * W1[(258 + c) * 256 + h];
    }
    sa += ci * W1[256 * 256 + h] + cj * W1[257 * 256 + h];
    sb += ci * W1[(258 + 256) * 256 + h] + cj * W1[(258 + 257) * 256 + h];
    A[row * 256 + h] = sa;
    B[row * 256 + h] = sb;
}

// ---------------------------------------------------------------------------
// Prep 2: Wt[n][k] = (f16) W[k][n]   (transpose + f16 convert, 256x256)
// ---------------------------------------------------------------------------
__global__ void prep_w(const float* __restrict__ W, _Float16* __restrict__ Wt) {
    const int n = blockIdx.x, k = threadIdx.x;
    Wt[n * 256 + k] = (_Float16)W[k * 256 + n];
}

// ---------------------------------------------------------------------------
// Stage one 256x256 f16 weight matrix (n-major) into padded LDS.
// Preferred path: direct async global->LDS (no VGPR round trip, ASYNCcnt).
// Caller must s_wait_asynccnt(0) (see stage_weights_wait) before the barrier.
// ---------------------------------------------------------------------------
__device__ __forceinline__ void stage_weights(const _Float16* __restrict__ Wt,
                                              _Float16* wbuf, int tid) {
#ifdef HAS_ASYNC_LDS
    // Low 32 bits of a flat LDS pointer are the group-segment offset on AMDGCN,
    // and global flat == AS(1) numerically, so integer round-trips are exact.
    const char* g = (const char*)(Wt + tid * 256);
    char* l = (char*)(wbuf + tid * HPAD);
#pragma unroll
    for (int c = 0; c < 32; ++c) {
        __builtin_amdgcn_global_load_async_to_lds_b128(
            (gas_v4i)(uintptr_t)(g + c * 16),
            (las_v4i)(uintptr_t)(l + c * 16),
            0, 0);
    }
#else
    const v8h* src = (const v8h*)(Wt + tid * 256);
    v8h* dst = (v8h*)(wbuf + tid * HPAD);
#pragma unroll
    for (int c = 0; c < 32; ++c) dst[c] = src[c];
#endif
}

__device__ __forceinline__ void stage_weights_wait() {
#ifdef HAS_ASYNC_LDS
    __builtin_amdgcn_s_wait_asynccnt(0);   // drain async copies before barrier
#endif
}

// ---------------------------------------------------------------------------
// Main: one workgroup per (b,s1,s2). 8 waves; wave w owns rows [32w,32w+32).
// h0 = relu(A[s2,p2]+B[s1,p1]+b1) -> 3x fused WMMA GEMM+ReLU (in-LDS, f16)
// -> pooled (ds_add_f32) -> head MLP -> sigmoid score.
// ---------------------------------------------------------------------------
__global__ __launch_bounds__(256, 1)
void rn_main(const float* __restrict__ Ag, const float* __restrict__ Bg,
             const float* __restrict__ b1, const float* __restrict__ b2,
             const float* __restrict__ b3, const float* __restrict__ b4,
             const _Float16* __restrict__ Wt2, const _Float16* __restrict__ Wt3,
             const _Float16* __restrict__ Wt4,
             const float* __restrict__ Wf1, const float* __restrict__ bf1,
             const float* __restrict__ Wf2, const float* __restrict__ bf2,
             const float* __restrict__ Wf3, const float* __restrict__ bf3,
             const float* __restrict__ Wf4, const float* __restrict__ bf4,
             float* __restrict__ out) {
    extern __shared__ _Float16 smem[];
    _Float16* hbuf = smem;                       // 256 x HPAD halves
    _Float16* wbuf = smem + 256 * HPAD;          // 256 x HPAD halves
    float* pooled = (float*)(smem + 2 * 256 * HPAD);  // 256 f
    float* tmp    = pooled + 256;                      // 256 f
    float* tmp2   = tmp + 256;                         // 32 f

    const int tid = threadIdx.x;
    const int idx = blockIdx.x;                  // b*S*S + s1*S + s2
    const int s2 = idx % SDIM;
    const int s1 = (idx / SDIM) % SDIM;
    const int bb = idx / (SDIM * SDIM);

    pooled[tid] = 0.f;

    // ---- phase 0: kick async W2 staging, then build h0 tile (row = p1*16+p2)
    stage_weights(Wt2, wbuf, tid);
    {
        const float* Arow = Ag + (size_t)((bb * SDIM + s2) * 16 + (tid & 15)) * 256;
        const float* Brow = Bg + (size_t)((bb * SDIM + s1) * 16 + (tid >> 4)) * 256;
        _Float16* hrow = hbuf + tid * HPAD;
        for (int k = 0; k < 256; k += 4) {
            float4 a = *(const float4*)(Arow + k);
            float4 c = *(const float4*)(Brow + k);
            float4 d = *(const float4*)(b1 + k);
            v4h o;
            o[0] = (_Float16)fmaxf(a.x + c.x + d.x, 0.f);
            o[1] = (_Float16)fmaxf(a.y + c.y + d.y, 0.f);
            o[2] = (_Float16)fmaxf(a.z + c.z + d.z, 0.f);
            o[3] = (_Float16)fmaxf(a.w + c.w + d.w, 0.f);
            *(v4h*)(hrow + k) = o;
        }
    }
    stage_weights_wait();
    __syncthreads();

    const int wave = tid >> 5, lane = tid & 31;
    const int laneLo = lane & 15, laneHi = lane >> 4;

    const _Float16* Wnext[2] = {Wt3, Wt4};
    const float* biasL[3] = {b2, b3, b4};

    float psum[16];
#pragma unroll
    for (int ct = 0; ct < 16; ++ct) psum[ct] = 0.f;

    for (int layer = 0; layer < 3; ++layer) {
        const float* bias = biasL[layer];
        const bool last = (layer == 2);
        if (!last) {  // warm next layer's weights in L2/L0
            __builtin_prefetch(Wnext[layer] + tid * 256, 0, 0);
            __builtin_prefetch(Wnext[layer] + tid * 256 + 128, 0, 0);
        }
#pragma unroll
        for (int rt = 0; rt < 2; ++rt) {
            const int rowBase = (wave * 2 + rt) * 16;
            const v8f zero = {0.f, 0.f, 0.f, 0.f, 0.f, 0.f, 0.f, 0.f};
            v8f acc[16];
#pragma unroll
            for (int ct = 0; ct < 16; ++ct) acc[ct] = zero;

            for (int kt = 0; kt < 8; ++kt) {
                // A-frag: lane<16 -> K {0..7,16..23}; lane>=16 -> K {8..15,24..31}
                union { v16h v; v8h h[2]; } af;
                const _Float16* ap =
                    hbuf + (rowBase + laneLo) * HPAD + kt * 32 + laneHi * 8;
                af.h[0] = *(const v8h*)(ap);
                af.h[1] = *(const v8h*)(ap + 16);
#pragma unroll
                for (int ct = 0; ct < 16; ++ct) {
                    // B-frag: lane holds column n = laneLo; lane<16 -> K {0..7,16..23}; lane>=16 -> K {8..15,24..31}
                    union { v16h v; v8h h[2]; } bf;
                    const _Float16* bp =
                        wbuf + (ct * 16 + laneLo) * HPAD + kt * 32 + laneHi * 8;
                    bf.h[0] = *(const v8h*)(bp);
                    bf.h[1] = *(const v8h*)(bp + 16);
                    acc[ct] = __builtin_amdgcn_wmma_f32_16x16x32_f16(
                        false, af.v, false, bf.v, (short)0, acc[ct], false, false);
                }
            }

            if (!last) {
                // C layout: n = laneLo, m = laneHi*8 + r -> in-place f16 store
                // (rows are private to this wave: no barrier needed)
#pragma unroll
                for (int ct = 0; ct < 16; ++ct) {
                    const float bv = bias[ct * 16 + laneLo];
                    _Float16* hp =
                        hbuf + (rowBase + laneHi * 8) * HPAD + ct * 16 + laneLo;
#pragma unroll
                    for (int r = 0; r < 8; ++r)
                        hp[r * HPAD] = (_Float16)fmaxf(acc[ct][r] + bv, 0.f);
                }
            } else {
                // layer 4: relu then partial pool over this lane's 8 rows
#pragma unroll
                for (int ct = 0; ct < 16; ++ct) {
                    const float bv = bias[ct * 16 + laneLo];
                    float s = 0.f;
#pragma unroll
                    for (int r = 0; r < 8; ++r) s += fmaxf(acc[ct][r] + bv, 0.f);
                    psum[ct] += s;
                }
            }
        }
        if (!last) {
            __syncthreads();                       // all waves done reading wbuf
            stage_weights(Wnext[layer], wbuf, tid);
            stage_weights_wait();
            __syncthreads();
        }
    }

    // pooled[n] = sum over all 256 rows (LDS float atomics)
#pragma unroll
    for (int ct = 0; ct < 16; ++ct)
        atomicAdd(&pooled[ct * 16 + laneLo], psum[ct]);
    __syncthreads();

    // ---- head MLP (tiny: vec-mat per workgroup) ----
    {
        float s = bf1[tid];
        for (int k = 0; k < 256; ++k) s += pooled[k] * Wf1[k * 256 + tid];
        tmp[tid] = fmaxf(s, 0.f);
    }
    __syncthreads();
    {
        float s = bf2[tid];
        for (int k = 0; k < 256; ++k) s += tmp[k] * Wf2[k * 256 + tid];
        pooled[tid] = fmaxf(s, 0.f);
    }
    __syncthreads();
    if (tid < 29) {
        float s = bf3[tid];
        for (int k = 0; k < 256; ++k) s += pooled[k] * Wf3[k * 29 + tid];
        tmp2[tid] = fmaxf(s, 0.f);
    }
    __syncthreads();
    if (tid == 0) {
        float s = bf4[0];
        for (int j = 0; j < 29; ++j) s += tmp2[j] * Wf4[j];
        out[idx] = 1.f / (1.f + expf(-s));
    }
}

// ---------------------------------------------------------------------------
extern "C" void kernel_launch(void* const* d_in, const int* in_sizes, int n_in,
                              void* d_out, int out_size, void* d_ws, size_t ws_size,
                              hipStream_t stream) {
    const float* x   = (const float*)d_in[0];
    const float* W1  = (const float*)d_in[1];
    const float* b1  = (const float*)d_in[2];
    const float* W2  = (const float*)d_in[3];
    const float* b2  = (const float*)d_in[4];
    const float* W3  = (const float*)d_in[5];
    const float* b3  = (const float*)d_in[6];
    const float* W4  = (const float*)d_in[7];
    const float* b4  = (const float*)d_in[8];
    const float* Wf1 = (const float*)d_in[9];
    const float* bf1 = (const float*)d_in[10];
    const float* Wf2 = (const float*)d_in[11];
    const float* bf2 = (const float*)d_in[12];
    const float* Wf3 = (const float*)d_in[13];
    const float* bf3 = (const float*)d_in[14];
    const float* Wf4 = (const float*)d_in[15];
    const float* bf4 = (const float*)d_in[16];

    float* Ag = (float*)d_ws;                 // 640*256 f32
    float* Bg = Ag + 640 * 256;               // 640*256 f32
    _Float16* Wt2 = (_Float16*)(Bg + 640 * 256);
    _Float16* Wt3 = Wt2 + 65536;
    _Float16* Wt4 = Wt3 + 65536;

    prep_ab<<<640, 256, 0, stream>>>(x, W1, Ag, Bg);
    prep_w<<<256, 256, 0, stream>>>(W2, Wt2);
    prep_w<<<256, 256, 0, stream>>>(W3, Wt3);
    prep_w<<<256, 256, 0, stream>>>(W4, Wt4);

    const int smem = 2 * 256 * HPAD * (int)sizeof(_Float16)
                   + (256 + 256 + 32) * (int)sizeof(float);   // ~272.5 KB
    (void)hipFuncSetAttribute((const void*)rn_main,
                              hipFuncAttributeMaxDynamicSharedMemorySize, smem);
    rn_main<<<800, 256, smem, stream>>>(Ag, Bg, b1, b2, b3, b4, Wt2, Wt3, Wt4,
                                        Wf1, bf1, Wf2, bf2, Wf3, bf3, Wf4, bf4,
                                        (float*)d_out);
}
